// MultiheadAttention_26036091748541
// MI455X (gfx1250) — matrix operand order repaired and checked by reading, not compiled, not run on hardware
//
#include <hip/hip_runtime.h>

// ---------------- problem constants ----------------
#define BZ   2
#define SQ   2048
#define DM   2048
#define HQN  16
#define HKV  4
#define HD   128
#define KVD  512
#define NREP 4          // HQN / HKV
#define NEG_INF (-3.4e38f)

typedef _Float16 v16h __attribute__((ext_vector_type(16)));
typedef _Float16 v8h  __attribute__((ext_vector_type(8)));
typedef float    v8f  __attribute__((ext_vector_type(8)));

// D = A(16x32) * B(32x16) + C, f16 in, f32 accum
#define WMMA_F16(a, b, c) \
  __builtin_amdgcn_wmma_f32_16x16x32_f16(false, (a), false, (b), (short)0, (c), false, false)

// A-fragment (16x32 f16): lane = row(lane&15); two contiguous 8-half runs at
// K = 8*(lane>>4) and 16 + 8*(lane>>4).  p must already include the first run.
__device__ __forceinline__ v16h load_a_frag(const _Float16* p) {
  v8h lo = *reinterpret_cast<const v8h*>(p);
  v8h hi = *reinterpret_cast<const v8h*>(p + 16);
  v16h r;
#pragma unroll
  for (int i = 0; i < 8; ++i) { r[i] = lo[i]; r[i + 8] = hi[i]; }
  return r;
}

// B-fragment (32x16 f16): lane = column(lane&15); two contiguous 8-half runs at
// K = 8*(lane>>4) and 16 + 8*(lane>>4).  p must already include the first run.
__device__ __forceinline__ v16h load_b_frag(const _Float16* p) {
  v8h lo = *reinterpret_cast<const v8h*>(p);
  v8h hi = *reinterpret_cast<const v8h*>(p + 16);
  v16h r;
#pragma unroll
  for (int i = 0; i < 8; ++i) { r[i] = lo[i]; r[i + 8] = hi[i]; }
  return r;
}

// ---------------- elementwise converts ----------------
__global__ __launch_bounds__(256) void k_f32_to_f16(
    const float* __restrict__ in, _Float16* __restrict__ out, int n) {
  int i = blockIdx.x * blockDim.x + threadIdx.x;
  int stride = gridDim.x * blockDim.x;
  for (; i < n; i += stride) out[i] = (_Float16)in[i];
}

// V16 [B, S, KVD] -> VT [B, KVD, S]
__global__ __launch_bounds__(256) void k_transpose_v(
    const _Float16* __restrict__ v, _Float16* __restrict__ vt, int n) {
  int i = blockIdx.x * blockDim.x + threadIdx.x;
  int stride = gridDim.x * blockDim.x;
  for (; i < n; i += stride) {
    int c = i % KVD;
    int s = (i / KVD) % SQ;
    int b = i / (KVD * SQ);
    vt[((size_t)b * KVD + c) * SQ + s] = v[i];
  }
}

// ---------------- GEMM: out[m,n] = sum_k A[m,k]*W[n,k] + bias[n] ----------------
// A:[M,K] f16 row-major, W:[N,K] f16 row-major. Block = 128 thr = 4 waves (2x2),
// each wave owns a 32x32 tile (2x2 WMMA accumulators).
template <bool OUT_F32>
__global__ __launch_bounds__(128, 1) void k_gemm_f16(
    const _Float16* __restrict__ A, const _Float16* __restrict__ W,
    const float* __restrict__ bias, void* __restrict__ out,
    int M, int N, int K) {
  const int lane = threadIdx.x & 31;
  const int wave = threadIdx.x >> 5;
  const int sub  = lane & 15;
  const int hf   = lane >> 4;

  const int bm = blockIdx.y * 64 + (wave >> 1) * 32;
  const int bn = blockIdx.x * 64 + (wave & 1) * 32;

  v8f acc[2][2] = {};

  const _Float16* a0 = A + (size_t)(bm + sub) * K + 8 * hf;
  const _Float16* a1 = a0 + (size_t)16 * K;
  const _Float16* b0 = W + (size_t)(bn + sub) * K + 8 * hf;
  const _Float16* b1 = b0 + (size_t)16 * K;

  for (int k = 0; k < K; k += 32) {
    v16h af0 = load_a_frag(a0 + k);
    v16h af1 = load_a_frag(a1 + k);
    v16h bf0 = load_b_frag(b0 + k);
    v16h bf1 = load_b_frag(b1 + k);
    acc[0][0] = WMMA_F16(af0, bf0, acc[0][0]);
    acc[0][1] = WMMA_F16(af0, bf1, acc[0][1]);
    acc[1][0] = WMMA_F16(af1, bf0, acc[1][0]);
    acc[1][1] = WMMA_F16(af1, bf1, acc[1][1]);
  }

#pragma unroll
  for (int j = 0; j < 2; ++j) {
    const int col = bn + j * 16 + sub;
    const float bval = bias ? bias[col] : 0.0f;
#pragma unroll
    for (int i = 0; i < 2; ++i) {
      const int rowbase = bm + i * 16 + 8 * hf;
#pragma unroll
      for (int r = 0; r < 8; ++r) {
        const float val = acc[i][j][r] + bval;
        if (OUT_F32)
          ((float*)out)[(size_t)(rowbase + r) * N + col] = val;
        else
          ((_Float16*)out)[(size_t)(rowbase + r) * N + col] = (_Float16)val;
      }
    }
  }
}

// ---------------- flash attention (GQA, causal, bias, key-pad mask) ----------------
// One wave handles one 16-query tile of one (b, h). Block = 4 independent waves.
__global__ __launch_bounds__(128, 1) void k_flash_attn(
    const _Float16* __restrict__ Q,    // [B,S,DM]
    const _Float16* __restrict__ Kp,   // [B,S,KVD]
    const _Float16* __restrict__ VT,   // [B,KVD,S]
    const float* __restrict__ bias,    // [HQN,S]
    const unsigned char* __restrict__ amask, // [B,S]
    _Float16* __restrict__ ctx) {      // [B,S,DM]
  __shared__ __align__(16) _Float16 psh[4][16 * 32];

  const int lane = threadIdx.x & 31;
  const int wave = threadIdx.x >> 5;
  const int sub  = lane & 15;
  const int hf   = lane >> 4;

  const int gw    = blockIdx.x * 4 + wave;
  const int qt    = gw & 127;          // S/16 tiles
  const int h     = (gw >> 7) & 15;
  const int b     = gw >> 11;
  const int qbase = qt * 16;
  const int hkv   = h / NREP;
  const float scale = 0.08838834764831845f;   // 1/sqrt(128)

  // Q fragments: 16 rows x 128 (d), four 16x32 A-frags kept in registers
  const _Float16* qp =
      Q + (size_t)(b * SQ + qbase + sub) * DM + h * HD + 8 * hf;
  v16h qf[4];
#pragma unroll
  for (int t = 0; t < 4; ++t) qf[t] = load_a_frag(qp + t * 32);

  v8f O[8] = {};
  float m_i[8], l_i[8];
#pragma unroll
  for (int r = 0; r < 8; ++r) { m_i[r] = NEG_INF; l_i[r] = 0.0f; }

  const _Float16* kbase = Kp + (size_t)b * SQ * KVD + hkv * HD + 8 * hf;
  const _Float16* vtb =
      VT + (size_t)(b * KVD + hkv * HD + sub) * SQ;   // + (t*16)*SQ per dd tile
  _Float16* pw = psh[wave];

  const int nkb = (qbase + 16 + 31) >> 5;   // causal: keys 0..qbase+15
  for (int kb = 0; kb < nkb; ++kb) {
    const int key0 = kb * 32;

    // ---- scores, subtile j computed fully (WMMA chain + epilogue) before
    //      subtile j+1 starts, to keep the live set small ----
    v8f s[2];
#pragma unroll
    for (int j = 0; j < 2; ++j) {
      v8f acc = {};
      const _Float16* kp = kbase + (size_t)(key0 + 16 * j + sub) * KVD;
#pragma unroll
      for (int t = 0; t < 4; ++t) {
        v16h bf = load_b_frag(kp + t * 32);
        acc = WMMA_F16(qf[t], bf, acc);
      }
      // scale + bias + padding mask + causal mask, immediately
      const int key = key0 + 16 * j + sub;
      const float bval = bias[h * SQ + key];
      const bool keep = amask[b * SQ + key] != 0;
#pragma unroll
      for (int r = 0; r < 8; ++r) {
        const int row = qbase + r + 8 * hf;
        float v = acc[r] * scale + bval;
        if (!keep || key > row) v = NEG_INF;
        acc[r] = v;
      }
      s[j] = acc;
    }

    // ---- online softmax (reduce across 16 lanes holding one row's keys) ----
#pragma unroll
    for (int r = 0; r < 8; ++r) {
      float vmax = fmaxf(s[0][r], s[1][r]);
      vmax = fmaxf(vmax, __shfl_xor(vmax, 1, 32));
      vmax = fmaxf(vmax, __shfl_xor(vmax, 2, 32));
      vmax = fmaxf(vmax, __shfl_xor(vmax, 4, 32));
      vmax = fmaxf(vmax, __shfl_xor(vmax, 8, 32));
      const float mn = fmaxf(m_i[r], vmax);
      const float p0 = __expf(s[0][r] - mn);
      const float p1 = __expf(s[1][r] - mn);
      float ps = p0 + p1;
      ps += __shfl_xor(ps, 1, 32);
      ps += __shfl_xor(ps, 2, 32);
      ps += __shfl_xor(ps, 4, 32);
      ps += __shfl_xor(ps, 8, 32);
      const float alpha = __expf(m_i[r] - mn);
      l_i[r] = l_i[r] * alpha + ps;
      m_i[r] = mn;
#pragma unroll
      for (int t = 0; t < 8; ++t) O[t][r] *= alpha;
      // scatter probabilities into A-layout staging tile (row-major 16x32)
      const int prow = r + 8 * hf;
      pw[prow * 32 + sub]      = (_Float16)p0;
      pw[prow * 32 + 16 + sub] = (_Float16)p1;
    }
    asm volatile("" ::: "memory");   // keep LDS scatter before the gather below

    // ---- O += P(16x32) . V(32x128) ----
    v16h pa = load_a_frag(pw + sub * 32 + 8 * hf);
#pragma unroll
    for (int t = 0; t < 8; ++t) {
      v16h bf = load_b_frag(vtb + (size_t)(t * 16) * SQ + key0 + 8 * hf);
      O[t] = WMMA_F16(pa, bf, O[t]);
    }
  }

  // ---- normalize + store context [b, s, h*HD + dd] ----
  _Float16* cp = ctx + (size_t)(b * SQ + qbase) * DM + h * HD;
#pragma unroll
  for (int t = 0; t < 8; ++t)
#pragma unroll
    for (int r = 0; r < 8; ++r)
      cp[(size_t)(r + 8 * hf) * DM + t * 16 + sub] =
          (_Float16)(O[t][r] / l_i[r]);
}

// ---------------- workspace layout (bytes) ----------------
#define OFF_Q16   ((size_t)0)            // B*S*D f16   = 16 MiB
#define OFF_KV16  ((size_t)16777216)     // 16 MiB
#define OFF_WQ16  ((size_t)33554432)     // 8 MiB
#define OFF_WK16  ((size_t)41943040)     // 2 MiB
#define OFF_WV16  ((size_t)44040192)     // 2 MiB
#define OFF_WO16  ((size_t)46137344)     // 8 MiB
#define OFF_QP    ((size_t)54525952)     // 16 MiB  (Q projection, f16)
#define OFF_KP    ((size_t)71303168)     // 4 MiB
#define OFF_VP    ((size_t)75497472)     // 4 MiB
#define OFF_VT    ((size_t)79691776)     // 4 MiB
#define OFF_CTX   ((size_t)83886080)     // 16 MiB  (total ~96 MiB)

extern "C" void kernel_launch(void* const* d_in, const int* in_sizes, int n_in,
                              void* d_out, int out_size, void* d_ws,
                              size_t ws_size, hipStream_t stream) {
  (void)in_sizes; (void)n_in; (void)out_size; (void)ws_size;

  const float* q    = (const float*)d_in[0];
  const float* kv   = (const float*)d_in[1];
  const float* abia = (const float*)d_in[2];
  const unsigned char* am = (const unsigned char*)d_in[3];
  const float* Wq = (const float*)d_in[4];
  const float* bq = (const float*)d_in[5];
  const float* Wk = (const float*)d_in[6];
  const float* bk = (const float*)d_in[7];
  const float* Wv = (const float*)d_in[8];
  const float* bv = (const float*)d_in[9];
  const float* Wo = (const float*)d_in[10];
  const float* bo = (const float*)d_in[11];

  char* ws = (char*)d_ws;
  _Float16* q16  = (_Float16*)(ws + OFF_Q16);
  _Float16* kv16 = (_Float16*)(ws + OFF_KV16);
  _Float16* wq16 = (_Float16*)(ws + OFF_WQ16);
  _Float16* wk16 = (_Float16*)(ws + OFF_WK16);
  _Float16* wv16 = (_Float16*)(ws + OFF_WV16);
  _Float16* wo16 = (_Float16*)(ws + OFF_WO16);
  _Float16* Qp   = (_Float16*)(ws + OFF_QP);
  _Float16* Kp   = (_Float16*)(ws + OFF_KP);
  _Float16* Vp   = (_Float16*)(ws + OFF_VP);
  _Float16* VT   = (_Float16*)(ws + OFF_VT);
  _Float16* CTX  = (_Float16*)(ws + OFF_CTX);

  const int nQ  = BZ * SQ * DM;     // 8,388,608
  const int nWq = DM * DM;          // 4,194,304
  const int nWk = KVD * DM;         // 1,048,576
  const int nV  = BZ * SQ * KVD;    // 2,097,152

  // f32 -> f16 conversions
  k_f32_to_f16<<<(nQ + 255) / 256, 256, 0, stream>>>(q, q16, nQ);
  k_f32_to_f16<<<(nQ + 255) / 256, 256, 0, stream>>>(kv, kv16, nQ);
  k_f32_to_f16<<<(nWq + 255) / 256, 256, 0, stream>>>(Wq, wq16, nWq);
  k_f32_to_f16<<<(nWk + 255) / 256, 256, 0, stream>>>(Wk, wk16, nWk);
  k_f32_to_f16<<<(nWk + 255) / 256, 256, 0, stream>>>(Wv, wv16, nWk);
  k_f32_to_f16<<<(nWq + 255) / 256, 256, 0, stream>>>(Wo, wo16, nWq);

  // projections: [B*S, K=2048] x [N, 2048]^T
  const int M = BZ * SQ;  // 4096
  k_gemm_f16<false><<<dim3(DM / 64, M / 64), 128, 0, stream>>>(
      q16, wq16, bq, Qp, M, DM, DM);
  k_gemm_f16<false><<<dim3(KVD / 64, M / 64), 128, 0, stream>>>(
      kv16, wk16, bk, Kp, M, KVD, DM);
  k_gemm_f16<false><<<dim3(KVD / 64, M / 64), 128, 0, stream>>>(
      kv16, wv16, bv, Vp, M, KVD, DM);

  // V transpose for P.V fragment loads
  k_transpose_v<<<(nV + 255) / 256, 256, 0, stream>>>(Vp, VT, nV);

  // flash attention: B*HQN*(S/16) = 4096 waves, 4 waves/block
  k_flash_attn<<<dim3(BZ * HQN * (SQ / 16) / 4), 128, 0, stream>>>(
      Qp, Kp, VT, abia, am, CTX);

  // output projection (f32 out)
  k_gemm_f16<true><<<dim3(DM / 64, M / 64), 128, 0, stream>>>(
      CTX, wo16, bo, d_out, M, DM, DM);
}
